// Flashback_82154134438719
// MI455X (gfx1250) — matrix operand order repaired and checked by reading, not compiled, not run on hardware
//
#include <hip/hip_runtime.h>
#include <hip/hip_bf16.h>

// ---------------------------------------------------------------------------
// Flashback forward for MI455X (gfx1250, wave32, WMMA, async-LDS + TDM)
//   T=64, B=256, H=256, V=10000, U=1024
//   d_out = y[T,B,V] f32 ++ h_final[B,H] f32
// bf16 WMMA (f32 accum) for GRU matmuls and the dominant
// [16384,512]x[512,10000] FC GEMM; f32 VALU for gates / ST weighting.
// FC GEMM staging: A tile via GLOBAL_LOAD_ASYNC_TO_LDS_B128 (ASYNCcnt),
// B tile via one TENSOR_LOAD_TO_LDS descriptor per k-chunk (TENSORcnt),
// double-buffered to overlap copy with WMMA.
// ---------------------------------------------------------------------------

#define TT   64
#define BB   256
#define HH   256
#define VV   10000
#define H3   768
#define K2H  512
#define TBM  (TT * BB)          // 16384 rows of the FC GEMM

typedef __bf16 bf16;
typedef __attribute__((ext_vector_type(16))) __bf16 v16bf;
typedef __attribute__((ext_vector_type(8)))  __bf16 v8bf;
typedef __attribute__((ext_vector_type(8)))  float  v8f;
typedef __attribute__((ext_vector_type(4)))  unsigned int u32x4;
typedef __attribute__((ext_vector_type(8)))  int          i32x8;
typedef __attribute__((ext_vector_type(4)))  int          i32x4;

#if __has_builtin(__builtin_amdgcn_tensor_load_to_lds)
#define FB_HAS_TDM 1
#else
#define FB_HAS_TDM 0
#endif

__device__ __forceinline__ v8f wmma_bf16(v16bf a, v16bf b, v8f c) {
  // 8 args: (neg_a, A, neg_b, B, c_mod, C, reuse_a, reuse_b)
  return __builtin_amdgcn_wmma_f32_16x16x32_bf16(
      false, a, false, b, (short)0, c, false, false);
}

__device__ __forceinline__ v8f v8f_zero() {
  v8f z;
#pragma unroll
  for (int e = 0; e < 8; ++e) z[e] = 0.0f;
  return z;
}

// --- CDNA5 async global->LDS copy (VGLOBAL encoding, tracked by ASYNCcnt) ---
// LDS byte address = low 32 bits of the flat shared pointer.
__device__ __forceinline__ void async_ld_b128(void* lds_dst, const void* gsrc) {
  unsigned int       l = (unsigned int)(unsigned long long)(uintptr_t)lds_dst;
  unsigned long long g = (unsigned long long)(uintptr_t)gsrc;
  asm volatile("global_load_async_to_lds_b128 %0, %1, off"
               :: "v"(l), "v"(g) : "memory");
}
__device__ __forceinline__ void wait_async0() {
  asm volatile("s_wait_asynccnt 0" ::: "memory");
}

// --- CDNA5 Tensor Data Mover: one 2D bf16 tile global->LDS (TENSORcnt). ---
// Descriptor per CDNA5 ISA (2D, groups 2/3 zero).
// tensor_d0/tensor_d1 are the REMAINING tensor extent from the tile origin, so
// rows/cols past the tensor read as zeros (hardware OOB rule).
__device__ __forceinline__ void tdm_load_2d_bf16(
    void* lds_dst, const void* gsrc,
    unsigned tensor_d0, unsigned tensor_d1,   // remaining elems (dim0) / rows (dim1)
    unsigned tile_d0, unsigned tile_d1,       // tile elems / rows
    unsigned d0_stride_elems,
    unsigned pad_interval_code, unsigned pad_amount_code, bool pad_en)
{
#if FB_HAS_TDM
  unsigned long long g = (unsigned long long)(uintptr_t)gsrc;
  unsigned int       l = (unsigned int)(uintptr_t)lds_dst;
  u32x4 g0;
  g0[0] = 1u;                                              // count=1, user D#
  g0[1] = l;                                               // lds_addr
  g0[2] = (unsigned)(g & 0xffffffffu);                     // global_addr[31:0]
  g0[3] = (unsigned)((g >> 32) & 0x01ffffffu) | (2u << 30);// addr[56:32] | type=2
  i32x8 g1;
  g1[0] = (int)((1u << 16)                                 // data_size = 2 bytes
                | (pad_en ? (1u << 20) : 0u)
                | (pad_interval_code << 22)
                | (pad_amount_code << 25));
  g1[1] = (int)((tensor_d0 & 0xffffu) << 16);              // dim0[15:0] @ [63:48]
  g1[2] = (int)(((tensor_d0 >> 16) & 0xffffu)
                | ((tensor_d1 & 0xffffu) << 16));          // dim0[31:16] | dim1[15:0]
  g1[3] = (int)(((tensor_d1 >> 16) & 0xffffu)
                | ((tile_d0 & 0xffffu) << 16));            // dim1[31:16] | tile0
  g1[4] = (int)(tile_d1 & 0xffffu);                        // tile1; tile2=0
  g1[5] = (int)d0_stride_elems;                            // dim0_stride[31:0]
  g1[6] = 0;                                               // stride[47:32]=0; dim1_stride lo=0
  g1[7] = 0;
  i32x4 z4; z4[0] = z4[1] = z4[2] = z4[3] = 0;
#if __clang_major__ >= 23
  i32x8 z8;
#pragma unroll
  for (int i = 0; i < 8; ++i) z8[i] = 0;
  __builtin_amdgcn_tensor_load_to_lds(g0, g1, z4, z4, z8, 0);
#else
  __builtin_amdgcn_tensor_load_to_lds(g0, g1, z4, z4, 0);
#endif
#else
  (void)lds_dst; (void)gsrc; (void)tensor_d0; (void)tensor_d1;
  (void)tile_d0; (void)tile_d1; (void)d0_stride_elems;
  (void)pad_interval_code; (void)pad_amount_code; (void)pad_en;
#endif
}

__device__ __forceinline__ void wait_tensor0() {
#if FB_HAS_TDM
  __builtin_amdgcn_s_wait_tensorcnt((short)0);
#endif
}

// A fragment (16x32 bf16): lanes 0-15 (M=lane) hold K{0..7},{16..23};
// lanes 16-31 (M=lane-16) hold K{8..15},{24..31}.
__device__ __forceinline__ v16bf load_frag_a(const bf16* tile, int stride, int lane) {
  int row  = lane & 15;
  int koff = (lane >> 4) * 8;
  const bf16* p = tile + row * stride + koff;
  union { v16bf v; v8bf h[2]; } u;
  u.h[0] = *(const v8bf*)(p);
  u.h[1] = *(const v8bf*)(p + 16);
  return u.v;
}

// B fragment (32x16 bf16), source stored [N][K] (K contiguous):
// lanes 0-15 (N=lane) hold K{0..7},{16..23}; lanes 16-31 hold K{8..15},{24..31}.
__device__ __forceinline__ v16bf load_frag_b(const bf16* tile, int stride, int lane) {
  int row  = lane & 15;
  int koff = (lane >> 4) * 8;
  const bf16* p = tile + row * stride + koff;
  union { v16bf v; v8bf h[2]; } u;
  u.h[0] = *(const v8bf*)(p);
  u.h[1] = *(const v8bf*)(p + 16);
  return u.v;
}

// ---------------------------------------------------------------------------
// K0: transpose+convert  src[K][N] f32 -> dst[N][K] bf16
// ---------------------------------------------------------------------------
__global__ void fb_transpose_bf16(const float* __restrict__ src,
                                  bf16* __restrict__ dst, int K, int N) {
  size_t idx   = (size_t)blockIdx.x * blockDim.x + threadIdx.x;
  size_t total = (size_t)K * N;
  if (idx < total) {
    int n = (int)(idx / K);
    int k = (int)(idx % K);
    dst[idx] = (bf16)src[(size_t)k * N + n];
  }
}

// ---------------------------------------------------------------------------
// K1: embedding + GRU. 16 blocks x 16 batch rows, 8 waves/block.
// ---------------------------------------------------------------------------
#define XP (HH + 8)   // padded LDS row stride (bf16 elems); 528B, 16B aligned

__global__ __launch_bounds__(256) void fb_gru(
    const int*   __restrict__ x,
    const float* __restrict__ h0,
    const float* __restrict__ poi_emb,
    const bf16*  __restrict__ wxT,   // [768][256] bf16
    const bf16*  __restrict__ whT,   // [768][256] bf16
    const float* __restrict__ gb,    // [768]
    float*       __restrict__ out,   // [T,B,H] f32
    float*       __restrict__ hfin)  // [B,H] f32 (tail of d_out)
{
  const int b0   = blockIdx.x * 16;
  const int tid  = threadIdx.x;
  const int lane = tid & 31;
  const int wave = tid >> 5;

  __shared__ __attribute__((aligned(16))) bf16  xe[16 * XP];
  __shared__ __attribute__((aligned(16))) bf16  hb[16 * XP];
  __shared__ __attribute__((aligned(16))) float hf[16 * XP];

  for (int idx = tid; idx < 16 * HH; idx += 256) {
    int r = idx >> 8, c = idx & 255;
    float v = h0[(b0 + r) * HH + c];
    hf[r * XP + c] = v;
    hb[r * XP + c] = (bf16)v;
  }
  __syncthreads();

  for (int t = 0; t < TT; ++t) {
    for (int idx = tid; idx < 16 * HH; idx += 256) {
      int r = idx >> 8, c = idx & 255;
      int poi = x[t * BB + (b0 + r)];
      xe[r * XP + c] = (bf16)poi_emb[(size_t)poi * HH + c];
    }
    __syncthreads();

    v8f accx[2][3], acch[2][3];
#pragma unroll
    for (int g = 0; g < 2; ++g)
#pragma unroll
      for (int m = 0; m < 3; ++m) { accx[g][m] = v8f_zero(); acch[g][m] = v8f_zero(); }

#pragma unroll 2
    for (int kk = 0; kk < HH; kk += 32) {
      v16bf ax = load_frag_a(xe + kk, XP, lane);
      v16bf ah = load_frag_a(hb + kk, XP, lane);
#pragma unroll
      for (int g = 0; g < 2; ++g) {
        int n_g = wave + 8 * g;
#pragma unroll
        for (int m = 0; m < 3; ++m) {            // r, z, n column triples
          int col0 = n_g * 16 + m * HH;          // row in wxT / whT
          v16bf bx = load_frag_b(wxT + (size_t)col0 * HH + kk, HH, lane);
          v16bf bh = load_frag_b(whT + (size_t)col0 * HH + kk, HH, lane);
          accx[g][m] = wmma_bf16(ax, bx, accx[g][m]);
          acch[g][m] = wmma_bf16(ah, bh, acch[g][m]);
        }
      }
    }
    __syncthreads();   // all WMMA reads of hb done before gate updates

    int coll = lane & 15;
    int rofs = (lane >> 4) * 8;
#pragma unroll
    for (int g = 0; g < 2; ++g) {
      int col = (wave + 8 * g) * 16 + coll;      // h column in [0,256)
      float b_r = gb[col], b_z = gb[col + HH], b_n = gb[col + 2 * HH];
#pragma unroll
      for (int e = 0; e < 8; ++e) {
        int row = e + rofs;                      // batch row in chunk
        float r = 1.0f / (1.0f + __expf(-(accx[g][0][e] + b_r + acch[g][0][e])));
        float z = 1.0f / (1.0f + __expf(-(accx[g][1][e] + b_z + acch[g][1][e])));
        float n = tanhf(accx[g][2][e] + b_n + r * acch[g][2][e]);
        float hp = hf[row * XP + col];
        float hn = (1.0f - z) * n + z * hp;
        hf[row * XP + col] = hn;
        hb[row * XP + col] = (bf16)hn;
        out[((size_t)t * BB + b0 + row) * HH + col] = hn;
      }
    }
    __syncthreads();
  }

  for (int idx = tid; idx < 16 * HH; idx += 256) {
    int r = idx >> 8, c = idx & 255;
    hfin[(b0 + r) * HH + c] = hf[r * XP + c];
  }
}

// ---------------------------------------------------------------------------
// K2: causal spatio-temporal weighting + concat, one wave per (i,b).
// ---------------------------------------------------------------------------
__global__ __launch_bounds__(256) void fb_stw(
    const float* __restrict__ t_in,
    const float* __restrict__ s_in,
    const float* __restrict__ out,      // [T,B,H] f32
    const int*   __restrict__ active_user,
    const float* __restrict__ user_emb,
    bf16*        __restrict__ A)        // [TBM][512] bf16
{
  __shared__ float wsh[8][TT];
  const int tid = threadIdx.x, lane = tid & 31, wave = tid >> 5;
  const int gw  = blockIdx.x * 8 + wave;        // (i,b) pair index
  const int i   = gw / BB, b = gw % BB;

  const float ti  = t_in[i * BB + b];
  const float si0 = s_in[(i * BB + b) * 2 + 0];
  const float si1 = s_in[(i * BB + b) * 2 + 1];
  const float W_T = 7.27220521664e-05f;   // 2*pi/86400
  float sum = 0.0f;
#pragma unroll
  for (int jj = 0; jj < 2; ++jj) {
    int j = lane + 32 * jj;
    float w = 0.0f;
    if (j <= i) {
      float dt = ti - t_in[j * BB + b];
      float d0 = si0 - s_in[(j * BB + b) * 2 + 0];
      float d1 = si1 - s_in[(j * BB + b) * 2 + 1];
      float ds = sqrtf(d0 * d0 + d1 * d1);
      float ft = 0.5f * (__cosf(dt * W_T) + 1.0f) * __expf(-(dt / 86400.0f) * 0.1f);
      float fs = __expf(-ds * 100.0f);
      w = ft * fs + 1e-10f;
    }
    wsh[wave][j] = w;
    sum += w;
  }
#pragma unroll
  for (int off = 16; off > 0; off >>= 1) sum += __shfl_xor(sum, off, 32);
  __syncthreads();

  float acc[8];
#pragma unroll
  for (int k = 0; k < 8; ++k) acc[k] = 0.0f;
  for (int j = 0; j <= i; ++j) {
    float w = wsh[wave][j];
    const float* op = out + ((size_t)j * BB + b) * HH;
#pragma unroll
    for (int k = 0; k < 8; ++k) acc[k] += w * op[lane + 32 * k];
  }
  float inv = 1.0f / sum;
  size_t row = (size_t)i * BB + b;
  int au = active_user[b];
#pragma unroll
  for (int k = 0; k < 8; ++k) {
    int hcol = lane + 32 * k;
    A[row * K2H + hcol]      = (bf16)(acc[k] * inv);
    A[row * K2H + HH + hcol] = (bf16)user_emb[(size_t)au * HH + hcol];
  }
}

// ---------------------------------------------------------------------------
// K3: FC GEMM  Y[16384,10000] = A @ fc_w + bias   (bf16 WMMA, f32 accum)
// 128x128 block tile, BK=32, 8 waves (4x2), each wave = 32x64 (2x4 tiles).
// A tile: per-lane GLOBAL_LOAD_ASYNC_TO_LDS_B128 (ASYNCcnt).
// B tile: one TENSOR_LOAD_TO_LDS descriptor per chunk (TENSORcnt), with LDS
// padding via the TDM pad fields (64B interval, 16B pad == LSTR row stride),
// and hardware OOB zero-fill past V.  Double-buffered copy/compute overlap.
// ---------------------------------------------------------------------------
#define BM 128
#define BN 128
#define BK 32
#define LSTR (BK + 8)   // 40 bf16 = 80B row, 16B aligned

__global__ __launch_bounds__(256) void fb_fc_gemm(
    const bf16*  __restrict__ A,     // [TBM][512]
    const bf16*  __restrict__ Bt,    // [V][512]
    const float* __restrict__ bias,  // [V]
    float*       __restrict__ Y)     // [TBM][V]
{
  __shared__ __attribute__((aligned(16))) bf16 sA[2][BM * LSTR];  // 2 x 10240 B
  __shared__ __attribute__((aligned(16))) bf16 sB[2][BN * LSTR];  // 2 x 10240 B

  const int tid = threadIdx.x, lane = tid & 31, wave = tid >> 5;
  const int bm = blockIdx.y * BM;
  const int bn = blockIdx.x * BN;
  const int wm = (wave & 3) * 32;          // wave row offset in tile
  const int wn = (wave >> 2) * 64;         // wave col offset in tile

  v8f acc[2][4];
#pragma unroll
  for (int m = 0; m < 2; ++m)
#pragma unroll
    for (int n = 0; n < 4; ++n) acc[m][n] = v8f_zero();

  const int lr  = tid >> 2;                // 0..63 (row group)
  const int seg = tid & 3;                 // 0..3  (8-bf16 segment of 32-K row)

  // Issue the async stage of one BK-chunk into LDS buffer `buf`.
  auto stage = [&](int buf, int kk) {
    // A tile: per-lane async b128 copies
#pragma unroll
    for (int p = 0; p < 2; ++p) {
      int row = lr + p * 64;
      const bf16* ga = A + (size_t)(bm + row) * K2H + kk + seg * 8;
      async_ld_b128(&sA[buf][row * LSTR + seg * 8], ga);
      if (kk + 2 * BK < K2H)
        __builtin_prefetch(ga + 2 * BK, 0, 3);    // -> global_prefetch_b8
    }
#if FB_HAS_TDM
    // B tile: one TDM descriptor (wave-level, EXEC-independent) from wave 0.
    // pad_interval code 3 -> pad every 16 DWORDs (64B = 32 bf16 = one K row);
    // pad_amount   code 3 -> insert 4 DWORDs (16B = 8 bf16) => LSTR row pitch.
    if (wave == 0) {
      tdm_load_2d_bf16(&sB[buf][0],
                       Bt + (size_t)bn * K2H + kk,
                       (unsigned)(K2H - kk),           // remaining K elems
                       (unsigned)(VV - bn),            // remaining N rows (OOB->0)
                       BK, BN, K2H,
                       3u, 3u, true);
    }
#else
    // Fallback: per-lane async copies for B as well.
#pragma unroll
    for (int p = 0; p < 2; ++p) {
      int row = lr + p * 64;
      int nrow = bn + row; if (nrow >= VV) nrow = VV - 1;   // clamp; masked at store
      const bf16* gbp = Bt + (size_t)nrow * K2H + kk + seg * 8;
      async_ld_b128(&sB[buf][row * LSTR + seg * 8], gbp);
    }
#endif
  };

  stage(0, 0);
  wait_async0();
  if (wave == 0) wait_tensor0();
  __syncthreads();

  int cur = 0;
  for (int kk = 0; kk < K2H; kk += BK) {
    const int nxt = cur ^ 1;
    if (kk + BK < K2H) stage(nxt, kk + BK);      // overlap copy with compute

    v16bf af[2], bfm[4];
    af[0] = load_frag_a(&sA[cur][(wm +  0) * LSTR], LSTR, lane);
    af[1] = load_frag_a(&sA[cur][(wm + 16) * LSTR], LSTR, lane);
#pragma unroll
    for (int n = 0; n < 4; ++n)
      bfm[n] = load_frag_b(&sB[cur][(wn + n * 16) * LSTR], LSTR, lane);
#pragma unroll
    for (int m = 0; m < 2; ++m)
#pragma unroll
      for (int n = 0; n < 4; ++n)
        acc[m][n] = wmma_bf16(af[m], bfm[n], acc[m][n]);

    wait_async0();                   // this wave's A copies into `nxt` landed
    if (wave == 0) wait_tensor0();   // wave 0's TDM B tile landed
    __syncthreads();                 // publish to all waves; `cur` free to reuse
    cur = nxt;
  }

  const int coll = lane & 15;
  const int rofs = (lane >> 4) * 8;
#pragma unroll
  for (int m = 0; m < 2; ++m)
#pragma unroll
    for (int n = 0; n < 4; ++n) {
      int col = bn + wn + n * 16 + coll;
      if (col < VV) {
        float bv = bias[col];
#pragma unroll
        for (int e = 0; e < 8; ++e) {
          int row = bm + wm + m * 16 + e + rofs;
          Y[(size_t)row * VV + col] = acc[m][n][e] + bv;
        }
      }
    }
}

// ---------------------------------------------------------------------------
extern "C" void kernel_launch(void* const* d_in, const int* in_sizes, int n_in,
                              void* d_out, int out_size, void* d_ws, size_t ws_size,
                              hipStream_t stream) {
  (void)in_sizes; (void)n_in; (void)out_size; (void)ws_size;
  const int*   x       = (const int*)  d_in[0];
  const float* t_in    = (const float*)d_in[1];
  const float* s_in    = (const float*)d_in[2];
  const float* h0      = (const float*)d_in[3];
  const int*   au      = (const int*)  d_in[4];
  const float* poi_emb = (const float*)d_in[5];
  const float* usr_emb = (const float*)d_in[6];
  const float* gru_wx  = (const float*)d_in[7];
  const float* gru_wh  = (const float*)d_in[8];
  const float* gru_b   = (const float*)d_in[9];
  const float* fc_w    = (const float*)d_in[10];
  const float* fc_b    = (const float*)d_in[11];

  float* y    = (float*)d_out;                       // [T,B,V]
  float* hfin = (float*)d_out + (size_t)TBM * VV;    // [B,H]

  char* ws = (char*)d_ws;
  size_t o = 0;
  float* gout = (float*)(ws + o); o += (size_t)TBM * HH * 4;     // GRU outputs f32
  bf16*  Abuf = (bf16*) (ws + o); o += (size_t)TBM * K2H * 2;    // GEMM A bf16
  bf16*  fcwT = (bf16*) (ws + o); o += (size_t)VV  * K2H * 2;    // fc_w^T bf16
  bf16*  wxT  = (bf16*) (ws + o); o += (size_t)H3 * HH * 2;      // wx^T bf16
  bf16*  whT  = (bf16*) (ws + o); o += (size_t)H3 * HH * 2;      // wh^T bf16

  {
    size_t n1 = (size_t)K2H * VV;
    fb_transpose_bf16<<<dim3((unsigned)((n1 + 255) / 256)), 256, 0, stream>>>(fc_w, fcwT, K2H, VV);
    size_t n2 = (size_t)HH * H3;
    fb_transpose_bf16<<<dim3((unsigned)((n2 + 255) / 256)), 256, 0, stream>>>(gru_wx, wxT, HH, H3);
    fb_transpose_bf16<<<dim3((unsigned)((n2 + 255) / 256)), 256, 0, stream>>>(gru_wh, whT, HH, H3);
  }
  fb_gru<<<dim3(16), 256, 0, stream>>>(x, h0, poi_emb, wxT, whT, gru_b, gout, hfin);
  fb_stw<<<dim3(TBM / 8), 256, 0, stream>>>(t_in, s_in, gout, au, usr_emb, Abuf);
  fb_fc_gemm<<<dim3((VV + BN - 1) / BN, TBM / BM), 256, 0, stream>>>(Abuf, fcwT, fc_b, y);
}
